// FeatureTransformerLayer_61426622267524
// MI455X (gfx1250) — matrix operand order repaired and checked by reading, not compiled, not run on hardware
//
#include <hip/hip_runtime.h>
#include <hip/hip_bf16.h>

// ---------------------------------------------------------------------------
// Fused single-query MHA for MI455X (gfx1250).
//   * Algebraic folding: K/V projections collapse through the single query /
//     attention vector: 18.5 GFLOP -> ~200 MFLOP, one streaming pass over the
//     35.6 MB kv tensor => HBM-bound (~1.5 us at 23.3 TB/s).
//   * bf16 WMMA (f32 accumulate) for all matmul stages except av = A @ kv (scalar f32 FMA).
//   * Tensor Data Mover (tensor_load_to_lds) streams kv into LDS (f32) with
//     TDM row padding (132-float pitch) to avoid LDS bank conflicts; waited
//     with s_wait_tensorcnt + workgroup barrier.
//   * Pass-1 LDS fragment reads are explicit float4 -> ds_load_b128 clauses
//     (round-2 asm showed serialized ds_load_2addr_b32 + per-pair waits).
// ---------------------------------------------------------------------------

typedef __attribute__((ext_vector_type(16))) __bf16 v16bf;
typedef __attribute__((ext_vector_type(8)))  float  v8f;
typedef unsigned int u32x4 __attribute__((ext_vector_type(4)));
typedef int          i32x4 __attribute__((ext_vector_type(4)));
typedef int          i32x8 __attribute__((ext_vector_type(8)));

#define B_   128
#define T_   32
#define N_   16
#define H_   128
#define NH_  4
#define L_   544                       // (N+1)*T
#define KVP  132                       // LDS row pitch in floats (TDM padding)
#define RSQRT_H 0.08838834764831845f   // 1/sqrt(128)

#if defined(__AMDGCN__) && __has_builtin(__builtin_amdgcn_tensor_load_to_lds)
#define HAVE_TDM 1
#endif

// ---- WMMA fragment helpers (wave32; A/B use contiguous k = 16*hi + j) ----------

// A (16x32 bf16, MxK): lane m=lane&15 holds row m; hi=lane>>4 selects the
// K half: element j -> K = kbase + hi*16 + j.
static __device__ inline v16bf load_a_frag(const float* A, int lda, int row0,
                                           int kbase, int lane) {
  const int r = lane & 15, hi = lane >> 4;
  const float* p = A + (size_t)(row0 + r) * lda;
  const int c0 = kbase + hi * 16;
  v16bf a;
#pragma unroll
  for (int j = 0; j < 8; ++j) {
    a[j]     = (__bf16)p[c0 + j];
    a[8 + j] = (__bf16)p[c0 + 8 + j];
  }
  return a;
}

// B (32x16 bf16, KxN): lane c=lane&15 holds column c; lanes 0-15 K=kbase..+15,
// lanes 16-31 K=kbase+16..+31, element j -> K = k0 + j.
// NT variant: memory holds B^T row-major (column contiguous).
static __device__ inline v16bf load_b_nt(const float* Bt, int ldb, int col0,
                                         int kbase, int lane) {
  const int c = lane & 15, hi = lane >> 4;
  const float* p = Bt + (size_t)(col0 + c) * ldb + kbase + hi * 16;
  v16bf b;
#pragma unroll
  for (int j = 0; j < 16; ++j) b[j] = (__bf16)p[j];
  return b;
}

// NN variant: memory holds B row-major [k][n] (column strided by ldb).
static __device__ inline v16bf load_b_nn(const float* Bm, int ldb, int col0,
                                         int kbase, int lane) {
  const int c = lane & 15, hi = lane >> 4;
  const float* p = Bm + (size_t)(kbase + hi * 16) * ldb + col0 + c;
  v16bf b;
#pragma unroll
  for (int j = 0; j < 16; ++j) b[j] = (__bf16)p[(size_t)j * ldb];
  return b;
}

static __device__ inline v8f wmma_bf16(v16bf a, v16bf b, v8f c) {
  // (neg_a, A, neg_b, B, c_mod, C, reuse_a, reuse_b)
  return __builtin_amdgcn_wmma_f32_16x16x32_bf16(false, a, false, b,
                                                 (short)0, c, false, false);
}

// D (16x16 f32): lane c=lane&15 is column, row = 8*(lane>>4) + v.
static __device__ inline void store_tile(float* D, int ldd, int row0, int col0,
                                         v8f acc, int lane) {
  const int c = lane & 15, hi = lane >> 4;
#pragma unroll
  for (int v = 0; v < 8; ++v)
    D[(size_t)(row0 + 8 * hi + v) * ldd + col0 + c] = acc[v];
}

#ifdef HAVE_TDM
// ---- Tensor Data Mover: 2-D tile load, global(f32) -> LDS, row padding ----
// D# per CDNA5 ISA ch.8: group0 {count=1, lds_addr, global_addr[56:0],
// type=2}; group1 {data_size=4B, pad_enable, pad_interval=6 (128 DWORDs),
// pad_amount=3 (4 DWORDs) -> 132-float LDS pitch, tensor/tile dims, stride};
// groups 2/3 zero (2-D tile: tile_dim2..4 = 0).
static __device__ inline void tdm_load_2d_f32(unsigned lds_addr,
                                              const float* gptr,
                                              unsigned rows, unsigned cols,
                                              unsigned stride_elems) {
  const unsigned long long ga = (unsigned long long)(size_t)gptr;
  u32x4 g0;
  g0[0] = 1u;                                     // count=1, user descriptor
  g0[1] = lds_addr;                               // LDS byte address
  g0[2] = (unsigned)(ga & 0xFFFFFFFFu);           // global_addr[31:0]
  g0[3] = (unsigned)((ga >> 32) & 0x01FFFFFFu)    // global_addr[56:32]
          | (2u << 30);                           // type = 2 ("image")
  i32x8 g1;
  g1[0] = (int)((2u << 16)                        // data_size = 4 bytes
                | (1u << 20)                      // pad_enable
                | (6u << 22)                      // pad_interval: 128 DWORDs
                | (3u << 25));                    // pad_amount: 4 DWORDs
  g1[1] = (int)((cols & 0xFFFFu) << 16);          // abar=0 | tensor_dim0 lo16
  g1[2] = (int)(((cols >> 16) & 0xFFFFu)          // tensor_dim0 hi16
                | ((rows & 0xFFFFu) << 16));      // tensor_dim1 lo16
  g1[3] = (int)(((rows >> 16) & 0xFFFFu)          // tensor_dim1 hi16
                | ((cols & 0xFFFFu) << 16));      // tile_dim0
  g1[4] = (int)(rows & 0xFFFFu);                  // tile_dim1 | tile_dim2=0
  g1[5] = (int)stride_elems;                      // tensor_dim0_stride lo32
  g1[6] = 0;                                      // stride hi16 | d1stride lo
  g1[7] = 0;                                      // d1stride hi
  const i32x4 z4 = {0, 0, 0, 0};
#if defined(__clang_major__) && __clang_major__ >= 23
  const i32x8 z8 = {0, 0, 0, 0, 0, 0, 0, 0};
  __builtin_amdgcn_tensor_load_to_lds(g0, g1, z4, z4, z8, 0);
#else
  __builtin_amdgcn_tensor_load_to_lds(g0, g1, z4, z4, 0);
#endif
}
#endif  // HAVE_TDM

// ---- Kernel 0: extract q[b,:] = node_enc[b, cts, :] ------------------------
__global__ void ft_extract_q(const float* __restrict__ node_enc,
                             const int* __restrict__ cts,
                             float* __restrict__ q) {
  const int i = blockIdx.x * blockDim.x + threadIdx.x;   // 16384 total
  const int b = i >> 7, h = i & 127;
  q[i] = node_enc[((size_t)b * T_ + cts[0]) * H_ + h];
}

// ---- Generic 128x128x128 batched WMMA GEMM (one wave per 16x16 tile) -------
template <bool BT>
__global__ void ft_gemm128(const float* __restrict__ Abase, long sA,
                           const float* __restrict__ Bbase, long sB,
                           float* __restrict__ Dbase, long sD) {
  const int lane = threadIdx.x & 31;
  const int tm = (blockIdx.x >> 3) * 16;
  const int tn = (blockIdx.x & 7) * 16;
  const int n  = blockIdx.y;
  const float* A  = Abase + (size_t)n * sA;
  const float* Bp = Bbase + (size_t)n * sB;
  float*       D  = Dbase + (size_t)n * sD;
  v8f acc = {0.f, 0.f, 0.f, 0.f, 0.f, 0.f, 0.f, 0.f};
#pragma unroll
  for (int kk = 0; kk < 4; ++kk) {
    v16bf a = load_a_frag(A, H_, tm, kk * 32, lane);
    v16bf b = BT ? load_b_nt(Bp, H_, tn, kk * 32, lane)
                 : load_b_nn(Bp, H_, tn, kk * 32, lane);
    acc = wmma_bf16(a, b, acc);
  }
  store_tile(D, H_, tm, tn, acc, lane);
}

// ---- Kernel 2: fused attention per batch element ---------------------------
// One workgroup per b (8 wave32s). TDM DMAs kv (f32, 132-float padded pitch)
// into LDS while the waves preload the qk^T WMMA B-fragments; then:
//   S tile = WMMA(kv_tile(16l x 32h) bf16, qk^T(32h x 16c; cols 0..3 = heads))
// masked softmax (wave per head), slf_attn output, and av = A @ kv (f32 LDS).
__global__ void ft_attention(const float* __restrict__ node_enc,
                             const float* __restrict__ neigh_enc,
                             const unsigned char* __restrict__ mask,
                             const float* __restrict__ qk,
                             float* __restrict__ av,
                             float* __restrict__ slf) {
  extern __shared__ char smem[];
  float* kv_lds = (float*)smem;                       // [544][132] f32 padded
  float* S      = (float*)(smem + (size_t)L_ * KVP * 4); // [4][544]

  const int b    = blockIdx.x;
  const int tid  = threadIdx.x;
  const int lane = tid & 31;
  const int wave = tid >> 5;

#ifdef HAVE_TDM
  // Issue the kv DMA from wave 0 only (TENSORcnt is per-wave; EXEC ignored).
  if (wave == 0) {
    const unsigned lds0 = (unsigned)(size_t)kv_lds;   // LDS byte offset
    tdm_load_2d_f32(lds0, node_enc + (size_t)b * T_ * H_, T_, H_, H_);
    tdm_load_2d_f32(lds0 + T_ * KVP * 4,
                    neigh_enc + (size_t)b * (N_ * T_) * H_, N_ * T_, H_, H_);
  }
#endif

  // Overlap with the DMA: preload B fragments of qk^T.
  // B[k=h][col]; cols 0..3 = heads, rest zero.
  v16bf bq[4];
  {
    const int c = lane & 15, hi = lane >> 4;
#pragma unroll
    for (int kk = 0; kk < 4; ++kk) {
      v16bf t;
      if (c < NH_) {
        const float* p = qk + ((size_t)c * B_ + b) * H_ + kk * 32 + hi * 16;
#pragma unroll
        for (int j = 0; j < 16; ++j) t[j] = (__bf16)p[j];
      } else {
#pragma unroll
        for (int j = 0; j < 16; ++j) t[j] = (__bf16)0.f;
      }
      bq[kk] = t;
    }
  }
  __builtin_prefetch(&mask[(size_t)b * L_ + tid], 0, 1);  // global_prefetch_b8

#ifdef HAVE_TDM
  if (wave == 0) __builtin_amdgcn_s_wait_tensorcnt(0);
#else
  // Fallback: cooperative float4 copy into the same padded LDS layout.
  for (int i = tid; i < L_ * (H_ / 4); i += 256) {
    const int l = i >> 5;            // 32 float4 per 128-float row
    const int c = (i & 31) * 4;
    const float4 v = (l < T_)
        ? *(const float4*)(node_enc + ((size_t)b * T_ + l) * H_ + c)
        : *(const float4*)(neigh_enc + ((size_t)b * (N_ * T_) + (l - T_)) * H_ + c);
    *(float4*)(kv_lds + (size_t)l * KVP + c) = v;
  }
#endif
  __syncthreads();

  // Pass 1: scores via WMMA; A-fragments read from padded LDS (conflict-free:
  // 132-float pitch shifts consecutive rows by 4 of the 64 LDS banks).
  // All 16 float4 row chunks are loaded up-front (ds_load_b128 clause, one
  // deep s_wait_dscnt) before the convert + WMMA chain.
  // 34 l-tiles over 8 waves; trip count uniform per wave => EXEC all-ones.
  const int r = lane & 15, hi = lane >> 4;
  for (int tile = wave; tile < L_ / 16; tile += 8) {
    const int l = tile * 16 + r;
    const float* rowp = kv_lds + (size_t)l * KVP;
    float4 f[16];
#pragma unroll
    for (int kk = 0; kk < 4; ++kk) {
      const int c0 = kk * 32 + hi * 16;   // 32-byte aligned chunk offsets
      f[kk * 4 + 0] = *(const float4*)(rowp + c0);
      f[kk * 4 + 1] = *(const float4*)(rowp + c0 + 4);
      f[kk * 4 + 2] = *(const float4*)(rowp + c0 + 8);
      f[kk * 4 + 3] = *(const float4*)(rowp + c0 + 12);
    }
    v8f acc = {0.f, 0.f, 0.f, 0.f, 0.f, 0.f, 0.f, 0.f};
#pragma unroll
    for (int kk = 0; kk < 4; ++kk) {
      v16bf a;
      const float4 x0 = f[kk * 4 + 0], x1 = f[kk * 4 + 1];
      const float4 y0 = f[kk * 4 + 2], y1 = f[kk * 4 + 3];
      a[0]  = (__bf16)x0.x; a[1]  = (__bf16)x0.y;
      a[2]  = (__bf16)x0.z; a[3]  = (__bf16)x0.w;
      a[4]  = (__bf16)x1.x; a[5]  = (__bf16)x1.y;
      a[6]  = (__bf16)x1.z; a[7]  = (__bf16)x1.w;
      a[8]  = (__bf16)y0.x; a[9]  = (__bf16)y0.y;
      a[10] = (__bf16)y0.z; a[11] = (__bf16)y0.w;
      a[12] = (__bf16)y1.x; a[13] = (__bf16)y1.y;
      a[14] = (__bf16)y1.z; a[15] = (__bf16)y1.w;
      acc = wmma_bf16(a, bq[kk], acc);
    }
    const int c = lane & 15;
    if (c < NH_) {
#pragma unroll
      for (int v = 0; v < 8; ++v) {
        const int ls = tile * 16 + 8 * hi + v;
        S[c * L_ + ls] = mask[(size_t)b * L_ + ls] ? -__builtin_inff()
                                                   : acc[v] * RSQRT_H;
      }
    }
  }
  __syncthreads();

  // Masked softmax over l: wave n owns head n (wave-local shuffle reduce).
  if (wave < NH_) {
    float* Sh = S + wave * L_;
    float m = -__builtin_inff();
    for (int l = lane; l < L_; l += 32) m = fmaxf(m, Sh[l]);
#pragma unroll
    for (int off = 16; off > 0; off >>= 1) m = fmaxf(m, __shfl_xor(m, off));
    float s = 0.f;
    for (int l = lane; l < L_; l += 32) {
      const float e = __expf(Sh[l] - m);   // exp(-inf) -> 0 for masked slots
      Sh[l] = e;
      s += e;
    }
#pragma unroll
    for (int off = 16; off > 0; off >>= 1) s += __shfl_xor(s, off);
    const float inv = 1.f / s;
    for (int l = lane; l < L_; l += 32) Sh[l] *= inv;
  }
  __syncthreads();

  // slf_attn[b, 0, l] = sum over heads of A.
  for (int l = tid; l < L_; l += 256)
    slf[(size_t)b * L_ + l] =
        S[l] + S[L_ + l] + S[2 * L_ + l] + S[3 * L_ + l];

  // Pass 2: av[n,b,h] = sum_l A[n,l] * kv[l,h]  (full-precision f32 kv).
  for (int o = tid; o < NH_ * H_; o += 256) {
    const int n = o >> 7, h = o & 127;
    float acc = 0.f;
    for (int l = 0; l < L_; ++l)
      acc += S[n * L_ + l] * kv_lds[(size_t)l * KVP + h];
    av[((size_t)n * B_ + b) * H_ + h] = acc;
  }
}

// ---- Kernel 3b: proj over 4 heads + bias + residual ------------------------
// x[b,j] = q[b,j] + prj_b[j] + sum_n sum_d ctx[n,b,d] * prj_w[j, n*128+d]
__global__ void ft_proj(const float* __restrict__ ctx,
                        const float* __restrict__ prj_w,
                        const float* __restrict__ prj_b,
                        const float* __restrict__ q,
                        float* __restrict__ x) {
  const int lane = threadIdx.x & 31;
  const int tm = (blockIdx.x >> 3) * 16;
  const int tn = (blockIdx.x & 7) * 16;
  v8f acc = {0.f, 0.f, 0.f, 0.f, 0.f, 0.f, 0.f, 0.f};
#pragma unroll
  for (int n = 0; n < NH_; ++n) {
#pragma unroll
    for (int kk = 0; kk < 4; ++kk) {
      v16bf a  = load_a_frag(ctx + (size_t)n * B_ * H_, H_, tm, kk * 32, lane);
      v16bf bb = load_b_nt(prj_w + n * H_, NH_ * H_, tn, kk * 32, lane);
      acc = wmma_bf16(a, bb, acc);
    }
  }
  const int c = lane & 15, hi = lane >> 4;
#pragma unroll
  for (int v = 0; v < 8; ++v) {
    const int row = tm + 8 * hi + v, col = tn + c;
    x[(size_t)row * H_ + col] = acc[v] + prj_b[col] + q[(size_t)row * H_ + col];
  }
}

// ---- Kernel 3c: layernorm (var = E[x^2] - mean^2, matching jnp.var) --------
__global__ void ft_layernorm(const float* __restrict__ x,
                             const float* __restrict__ gamma,
                             const float* __restrict__ beta,
                             float* __restrict__ out) {
  const int b = blockIdx.x, t = threadIdx.x;
  const int lane = t & 31, wave = t >> 5;
  const float v = x[(size_t)b * H_ + t];
  float s = v, s2 = v * v;
#pragma unroll
  for (int off = 16; off > 0; off >>= 1) {
    s  += __shfl_xor(s, off);
    s2 += __shfl_xor(s2, off);
  }
  __shared__ float sb[4], s2b[4];
  if (lane == 0) { sb[wave] = s; s2b[wave] = s2; }
  __syncthreads();
  s  = sb[0] + sb[1] + sb[2] + sb[3];
  s2 = s2b[0] + s2b[1] + s2b[2] + s2b[3];
  const float mean = s * (1.f / H_);
  const float var  = s2 * (1.f / H_) - mean * mean;
  out[(size_t)b * H_ + t] =
      (v - mean) * rsqrtf(var + 1e-6f) * gamma[t] + beta[t];
}

// ---------------------------------------------------------------------------
extern "C" void kernel_launch(void* const* d_in, const int* in_sizes, int n_in,
                              void* d_out, int out_size, void* d_ws,
                              size_t ws_size, hipStream_t stream) {
  (void)in_sizes; (void)n_in; (void)out_size; (void)ws_size;
  const float*         node_enc  = (const float*)d_in[0];
  const float*         neigh_enc = (const float*)d_in[1];
  const unsigned char* attn_mask = (const unsigned char*)d_in[2]; // jnp bool_
  const int*           cts       = (const int*)d_in[3];
  const float*         w_qs      = (const float*)d_in[4];
  const float*         w_ks      = (const float*)d_in[5];
  const float*         w_vs      = (const float*)d_in[6];
  const float*         prj_w     = (const float*)d_in[7];
  const float*         prj_b     = (const float*)d_in[8];
  const float*         ln_g      = (const float*)d_in[9];
  const float*         ln_b      = (const float*)d_in[10];

  float* out = (float*)d_out;      // [128][128] layernorm output
  float* slf = out + B_ * H_;      // [128][544] head-summed attention

  // Workspace carve (f32): q, qs, qk, av, ctx, x  (~1.13 MB total)
  float* ws  = (float*)d_ws;
  float* qws = ws;                       // 16384
  float* qs  = qws + B_ * H_;            // 4*16384
  float* qk  = qs  + NH_ * B_ * H_;      // 4*16384
  float* av  = qk  + NH_ * B_ * H_;      // 4*16384
  float* ctx = av  + NH_ * B_ * H_;      // 4*16384
  float* xws = ctx + NH_ * B_ * H_;      // 16384

  ft_extract_q<<<64, 256, 0, stream>>>(node_enc, cts, qws);
  // qs[n] = q @ w_qs[n]             (B row-major over output dim -> NN)
  ft_gemm128<false><<<dim3(64, NH_), 32, 0, stream>>>(
      qws, 0, w_qs, H_ * H_, qs, B_ * H_);
  // qk[n] = qs[n] @ w_ks[n]^T       (contract over d -> NT, contiguous B)
  ft_gemm128<true><<<dim3(64, NH_), 32, 0, stream>>>(
      qs, B_ * H_, w_ks, H_ * H_, qk, B_ * H_);

  // LDS: padded kv (544*132 f32) + scores (4*544 f32) = 295,936 B <= 320 KB.
  const size_t smem = ((size_t)L_ * KVP + (size_t)NH_ * L_) * sizeof(float);
  ft_attention<<<B_, 256, smem, stream>>>(node_enc, neigh_enc, attn_mask, qk,
                                          av, slf);

  // ctx[n] = av[n] @ w_vs[n]        (NN)
  ft_gemm128<false><<<dim3(64, NH_), 32, 0, stream>>>(
      av, B_ * H_, w_vs, H_ * H_, ctx, B_ * H_);

  ft_proj<<<64, 32, 0, stream>>>(ctx, prj_w, prj_b, qws, xws);
  ft_layernorm<<<B_, H_, 0, stream>>>(xws, ln_g, ln_b, out);
}
